// MambaVortex_TFV_32384053412149
// MI455X (gfx1250) — compile-verified, matrix operand order checked by reading, not run on hardware
//
#include <hip/hip_runtime.h>
#include <hip/hip_bf16.h>

typedef _Float16 half_t;
typedef __attribute__((ext_vector_type(16))) _Float16 v16h;
typedef __attribute__((ext_vector_type(8)))  float    v8f;

#define NB 1024   // batch
#define SS 29     // seq
#define DD 128    // channels
#define NN 30     // nodes after concat
#define NF 15     // rfft bins
#define MH 38     // mlp hidden

// ---------------- small math helpers ----------------
__device__ __forceinline__ float lrelu_(float x){ return x > 0.f ? x : 0.2f * x; }
__device__ __forceinline__ float elu_  (float x){ return x > 0.f ? x : (__expf(x) - 1.f); }
__device__ __forceinline__ float sigm_ (float x){ return 1.f / (1.f + __expf(-x)); }

// ---------------- WMMA fragment loaders (wave32, 16x16x32 f16) ----------------
// A 16x32 f16 layout: lane<16 -> M=lane, K-half 0; lane>=16 -> K-half 8.
__device__ __forceinline__ v16h frag_a_f32(const float* A, int lda, int m0, int k0, int lane)
{
  int m  = m0 + (lane & 15);
  int kb = k0 + ((lane >> 4) << 3);
  const float* p = A + m * lda + kb;
  v16h a;
#pragma unroll
  for (int i = 0; i < 4; ++i) {
    a[2*i]     = (half_t)p[2*i];
    a[2*i+1]   = (half_t)p[2*i+1];
    a[8+2*i]   = (half_t)p[16+2*i];
    a[8+2*i+1] = (half_t)p[16+2*i+1];
  }
  return a;
}
// B 32x16 (K x N), f32 source
__device__ __forceinline__ v16h frag_b_f32(const float* B, int ldb, int k0, int n0, int lane)
{
  int n  = n0 + (lane & 15);
  int kb = k0 + ((lane >> 4) << 3);
  v16h b;
#pragma unroll
  for (int i = 0; i < 4; ++i) {
    b[2*i]     = (half_t)B[(kb + 2*i    ) * ldb + n];
    b[2*i+1]   = (half_t)B[(kb + 2*i + 1) * ldb + n];
    b[8+2*i]   = (half_t)B[(kb + 16 + 2*i    ) * ldb + n];
    b[8+2*i+1] = (half_t)B[(kb + 16 + 2*i + 1) * ldb + n];
  }
  return b;
}
// B = X^T : logical B[k][n] = X[n][k]  (for Gram matrices X X^T)
__device__ __forceinline__ v16h frag_bt_f32(const float* X, int ldx, int k0, int n0, int lane)
{
  int n  = n0 + (lane & 15);
  int kb = k0 + ((lane >> 4) << 3);
  const float* p = X + n * ldx + kb;
  v16h b;
#pragma unroll
  for (int i = 0; i < 4; ++i) {
    b[2*i]     = (half_t)p[2*i];
    b[2*i+1]   = (half_t)p[2*i+1];
    b[8+2*i]   = (half_t)p[16+2*i];
    b[8+2*i+1] = (half_t)p[16+2*i+1];
  }
  return b;
}
// B from f16 (pre-staged zero-padded weights in LDS)
__device__ __forceinline__ v16h frag_b_h(const half_t* B, int ldb, int k0, int n0, int lane)
{
  int n  = n0 + (lane & 15);
  int kb = k0 + ((lane >> 4) << 3);
  v16h b;
#pragma unroll
  for (int i = 0; i < 4; ++i) {
    b[2*i]     = B[(kb + 2*i    ) * ldb + n];
    b[2*i+1]   = B[(kb + 2*i + 1) * ldb + n];
    b[8+2*i]   = B[(kb + 16 + 2*i    ) * ldb + n];
    b[8+2*i+1] = B[(kb + 16 + 2*i + 1) * ldb + n];
  }
  return b;
}

// single 16x16 output tile, K accumulated (compile-time K -> unrolled WMMA chain)
template<int K>
__device__ __forceinline__ v8f wmma_tile(const float* A, int lda, const float* B, int ldb,
                                         int bt, int m0, int n0, int lane)
{
  v8f c = {};
#pragma unroll
  for (int k0 = 0; k0 < K; k0 += 32) {
    v16h a = frag_a_f32(A, lda, m0, k0, lane);
    v16h b = bt ? frag_bt_f32(B, ldb, k0, n0, lane) : frag_b_f32(B, ldb, k0, n0, lane);
    c = __builtin_amdgcn_wmma_f32_16x16x32_f16(false, a, false, b, (short)0, c, false, false);
  }
  return c;
}

// fully-unrolled cooperative GEMM: 8 waves round-robin over 16x16 tiles
template<int M, int N, int K, int NW = 8>
__device__ __forceinline__ void wmma_gemm(const float* A, int lda, const float* B, int ldb, int bt,
                                          float* C, int ldc, int wave, int lane)
{
  constexpr int NT = N / 16;
  constexpr int TILES = (M / 16) * NT;
#pragma unroll
  for (int t0 = 0; t0 < TILES; t0 += NW) {
    int t = t0 + wave;
    if (t < TILES) {
      int m0 = (t / NT) << 4, n0 = (t % NT) << 4;
      v8f c = wmma_tile<K>(A, lda, B, ldb, bt, m0, n0, lane);
      int col = n0 + (lane & 15);
      int rb  = m0 + ((lane >> 4) << 3);
#pragma unroll
      for (int r = 0; r < 8; ++r) C[(rb + r) * ldc + col] = c[r];
    }
  }
}

template<int M, int N, int K, int NW = 8>
__device__ __forceinline__ void wmma_gemm_h(const float* A, int lda, const half_t* B, int ldb,
                                            float* C, int ldc, int wave, int lane)
{
  constexpr int NT = N / 16;
  constexpr int TILES = (M / 16) * NT;
#pragma unroll
  for (int t0 = 0; t0 < TILES; t0 += NW) {
    int t = t0 + wave;
    if (t < TILES) {
      int m0 = (t / NT) << 4, n0 = (t % NT) << 4;
      v8f c = {};
#pragma unroll
      for (int k0 = 0; k0 < K; k0 += 32) {
        v16h a = frag_a_f32(A, lda, m0, k0, lane);
        v16h b = frag_b_h(B, ldb, k0, n0, lane);
        c = __builtin_amdgcn_wmma_f32_16x16x32_f16(false, a, false, b, (short)0, c, false, false);
      }
      int col = n0 + (lane & 15);
      int rb  = m0 + ((lane >> 4) << 3);
#pragma unroll
      for (int r = 0; r < 8; ++r) C[(rb + r) * ldc + col] = c[r];
    }
  }
}

// ---------------- utility ----------------
__global__ void k_zero(float* p, int n)
{
  int i = blockIdx.x * blockDim.x + threadIdx.x;
  if (i < n) p[i] = 0.f;
}

// ---------------- stage 1: encoder (PE + 2x spectral block + FCA + LN + mean row) ----------------
__global__ __launch_bounds__(256)
void k_encoder(const float* __restrict__ x,
               const float* cwr, const float* cwi,
               const float* ln1g, const float* ln1b,
               const float* ln2g, const float* ln2b,
               const float* W1, const float* b1,
               const float* W2, const float* b2,
               const float* fw1, const float* fb1,
               const float* fw2, const float* fb2,
               const float* lng, const float* lnb,
               float* __restrict__ hout)
{
  __shared__ float  xb[SS * DD];      // residual stream [29][128]
  __shared__ float  hb[32 * DD];      // normalized / padded [32][128]
  __shared__ float  fbuf[30 * DD];    // Xr/Xi (15*128 each) or hmid [32][64]
  __shared__ half_t gh[8192];         // staged f16 weights
  __shared__ float  st[256];          // stats

  const int tid = threadIdx.x, lane = tid & 31, wave = tid >> 5;
  const int b = blockIdx.x;
  const float* xg = x + (size_t)b * SS * DD;

  // pull MLP weights toward this WGP while we do PE / LN / DFT
  __builtin_prefetch(W1 + (tid << 4), 0, 0);
  __builtin_prefetch(W2 + (tid << 4), 0, 0);

  // load + positional encoding
  for (int i = tid; i < SS * DD; i += 256) {
    int t = i >> 7, d = i & 127;
    float freq = __powf(10000.0f, (float)(d & ~1) / (float)DD);
    float ang  = (float)t / freq;
    float pe   = (d & 1) ? __cosf(ang) : __sinf(ang);
    xb[i] = xg[i] + pe;
  }
  for (int i = tid; i < 32 * DD; i += 256) hb[i] = 0.f;
  __syncthreads();

  for (int blk = 0; blk < 2; ++blk) {
    const float* g1  = ln1g + blk * DD;  const float* bb1_ = ln1b + blk * DD;
    const float* g2  = ln2g + blk * DD;  const float* bb2_ = ln2b + blk * DD;
    const float* w1g = W1  + blk * DD * MH;   const float* b1g = b1 + blk * MH;
    const float* w2g = W2  + blk * MH * DD;   const float* b2g = b2 + blk * DD;
    const float* wr  = cwr + blk * NF * DD;   const float* wi  = cwi + blk * NF * DD;

    // LN1 stats
    if (tid < SS) {
      float m = 0.f;
      for (int d = 0; d < DD; ++d) m += xb[tid * DD + d];
      m *= (1.f / DD);
      float v = 0.f;
      for (int d = 0; d < DD; ++d) { float t0 = xb[tid * DD + d] - m; v += t0 * t0; }
      v *= (1.f / DD);
      st[tid] = m; st[32 + tid] = rsqrtf(v + 1e-5f);
    }
    __syncthreads();
    for (int i = tid; i < SS * DD; i += 256) {
      int t = i >> 7, d = i & 127;
      hb[i] = (xb[i] - st[t]) * st[32 + t] * g1[d] + bb1_[d];
    }
    __syncthreads();

    // rDFT over time (29 -> 15 bins), complex weight multiply
    float* Xr = fbuf; float* Xi = fbuf + NF * DD;
    for (int i = tid; i < NF * DD; i += 256) {
      int f = i >> 7, d = i & 127;
      float w = 6.2831853072f * (float)f / (float)SS;
      float ar = 0.f, ai = 0.f;
      for (int t = 0; t < SS; ++t) {
        float sv, cv; __sincosf(w * (float)t, &sv, &cv);
        float v = hb[t * DD + d];
        ar += v * cv; ai -= v * sv;
      }
      float r = wr[i], q = wi[i];
      Xr[i] = ar * r - ai * q;
      Xi[i] = ar * q + ai * r;
    }
    __syncthreads();
    // irDFT, residual add
    for (int i = tid; i < SS * DD; i += 256) {
      int t = i >> 7, d = i & 127;
      float acc = Xr[d];
      for (int f = 1; f < NF; ++f) {
        float w = 6.2831853072f * (float)f * (float)t / (float)SS;
        float sv, cv; __sincosf(w, &sv, &cv);
        acc += 2.f * (Xr[f * DD + d] * cv - Xi[f * DD + d] * sv);
      }
      xb[i] += acc * (1.f / (float)SS);
    }
    __syncthreads();

    // LN2 -> hb (MLP input, padded rows stay 0)
    if (tid < SS) {
      float m = 0.f;
      for (int d = 0; d < DD; ++d) m += xb[tid * DD + d];
      m *= (1.f / DD);
      float v = 0.f;
      for (int d = 0; d < DD; ++d) { float t0 = xb[tid * DD + d] - m; v += t0 * t0; }
      v *= (1.f / DD);
      st[tid] = m; st[32 + tid] = rsqrtf(v + 1e-5f);
    }
    __syncthreads();
    for (int i = tid; i < SS * DD; i += 256) {
      int t = i >> 7, d = i & 127;
      hb[i] = (xb[i] - st[t]) * st[32 + t] * g2[d] + bb2_[d];
    }
    __syncthreads();

    // MLP gemm1: [32,128] x [128,48] (38 padded) via WMMA
    for (int i = tid; i < 128 * 48; i += 256) {
      int k = i / 48, n = i % 48;
      gh[i] = (half_t)((n < MH) ? w1g[k * MH + n] : 0.f);
    }
    float* hm = fbuf; // [32][64], cols >=48 zero
    for (int i = tid; i < 32 * 64; i += 256) hm[i] = 0.f;
    __syncthreads();
    wmma_gemm_h<32, 48, 128>(hb, DD, gh, 48, hm, 64, wave, lane);
    __syncthreads();
    for (int i = tid; i < SS * 48; i += 256) {
      int r = i / 48, n = i % 48;
      float v = hm[r * 64 + n] + ((n < MH) ? b1g[n] : 0.f);
      hm[r * 64 + n] = fmaxf(v, 0.f);
    }
    __syncthreads();
    // gemm2: [32,64] x [64,128] (38 padded K)
    for (int i = tid; i < 64 * DD; i += 256) {
      int k = i >> 7, n = i & 127;
      gh[i] = (half_t)((k < MH) ? w2g[k * DD + n] : 0.f);
    }
    __syncthreads();
    wmma_gemm_h<32, 128, 64>(hm, 64, gh, DD, hb, DD, wave, lane);
    __syncthreads();
    for (int i = tid; i < SS * DD; i += 256) xb[i] += hb[i] + b2g[i & 127];
    __syncthreads();
  }

  // FCA: channel attention over the 29 time rows
  if (tid < SS) {
    float s = 0.f;
    for (int d = 0; d < DD; ++d) s += xb[tid * DD + d];
    st[64 + tid] = s * (1.f / DD);
  }
  __syncthreads();
  if (tid < 8) {
    float s = fb1[tid];
    for (int t = 0; t < SS; ++t) s += st[64 + t] * fw1[t * 8 + tid];
    st[96 + tid] = fmaxf(s, 0.f);
  }
  __syncthreads();
  if (tid < SS) {
    float s = fb2[tid];
    for (int j = 0; j < 8; ++j) s += st[96 + j] * fw2[j * SS + tid];
    st[104 + tid] = sigm_(s);
  }
  __syncthreads();
  for (int i = tid; i < SS * DD; i += 256) xb[i] *= (1.f + st[104 + (i >> 7)]);
  __syncthreads();

  // final LN + mean row -> hout [30][128]
  if (tid < SS) {
    float m = 0.f;
    for (int d = 0; d < DD; ++d) m += xb[tid * DD + d];
    m *= (1.f / DD);
    float v = 0.f;
    for (int d = 0; d < DD; ++d) { float t0 = xb[tid * DD + d] - m; v += t0 * t0; }
    v *= (1.f / DD);
    st[tid] = m; st[32 + tid] = rsqrtf(v + 1e-5f);
  }
  __syncthreads();
  for (int i = tid; i < SS * DD; i += 256) {
    int t = i >> 7, d = i & 127;
    hb[i] = (xb[i] - st[t]) * st[32 + t] * lng[d] + lnb[d];
  }
  __syncthreads();
  float* hg = hout + (size_t)b * NN * DD;
  for (int i = tid; i < SS * DD; i += 256) hg[i] = hb[i];
  if (tid < DD) {
    float m = 0.f;
    for (int t = 0; t < SS; ++t) m += hb[t * DD + tid];
    hg[SS * DD + tid] = m * (1.f / SS);
  }
}

// ---------------- stage 2: hypergraph (30 nodes, k=5) with WMMA Gram + messages ----------------
__global__ __launch_bounds__(256)
void k_hyper30(const float* __restrict__ hin, const float* W, const float* bias,
               float* __restrict__ htout, float* cnt)
{
  __shared__ float h[32 * DD];
  __shared__ float xnm[32 * DD];   // xn, then messages m
  __shared__ float th[32 * DD];    // theta, then out
  __shared__ float Ss[32 * 32];
  __shared__ float Hr[32 * 32];    // raw membership (edge x node)
  __shared__ float Hs[32 * 32];    // scaled (and later transposed-scaled)
  __shared__ float aux[128];       // De[32], Dv[32], norms[32]

  const int tid = threadIdx.x, lane = tid & 31, wave = tid >> 5;
  const int b = blockIdx.x;
  const float* hgl = hin + (size_t)b * NN * DD;

  __builtin_prefetch(W + (tid << 6), 0, 0);

  for (int i = tid; i < 32 * DD; i += 256) h[i] = (i < NN * DD) ? hgl[i] : 0.f;
  __syncthreads();
  if (tid < NN) {
    float s = 1e-8f;
    for (int d = 0; d < DD; ++d) { float v = h[tid * DD + d]; s += v * v; }
    aux[64 + tid] = rsqrtf(s);
  }
  __syncthreads();
  for (int i = tid; i < 32 * DD; i += 256) {
    int r = i >> 7;
    xnm[i] = (r < NN) ? h[i] * aux[64 + r] : 0.f;
  }
  __syncthreads();
  // Gram: Ssim = xn xn^T  (WMMA, transposed-B)
  wmma_gemm<32, 32, 128>(xnm, DD, xnm, DD, 1, Ss, 32, wave, lane);
  __syncthreads();
  // top-5 per edge row
  if (tid < 32) {
    if (tid < NN) {
      float prev = 1e30f;
      for (int p = 0; p < 5; ++p) {
        float best = -1e30f;
        for (int n = 0; n < NN; ++n) {
          float v = Ss[tid * 32 + n];
          if (v < prev && v > best) best = v;
        }
        prev = best;
      }
      for (int n = 0; n < 32; ++n)
        Hr[tid * 32 + n] = (n < NN && Ss[tid * 32 + n] >= prev) ? 1.f : 0.f;
    } else {
      for (int n = 0; n < 32; ++n) Hr[tid * 32 + n] = 0.f;
    }
  }
  __syncthreads();
  if (tid < 32) {            // De
    float s = 0.f;
    for (int n = 0; n < NN; ++n) s += Hr[tid * 32 + n];
    aux[tid] = s;
  }
  __syncthreads();
  for (int i = tid; i < 32 * 32; i += 256) Hs[i] = Hr[i] / (aux[i >> 5] + 1e-8f);
  // theta = h @ W + bias
  wmma_gemm<32, 128, 128>(h, DD, W, DD, 0, th, DD, wave, lane);
  __syncthreads();
  for (int i = tid; i < 32 * DD; i += 256) th[i] += bias[i & 127];
  __syncthreads();
  // m = (H/De) @ theta  -> xnm
  wmma_gemm<32, 128, 32>(Hs, 32, th, DD, 0, xnm, DD, wave, lane);
  __syncthreads();
  if (tid < 32) {            // Dv
    float s = 0.f;
    for (int e = 0; e < NN; ++e) s += Hr[e * 32 + tid];
    aux[32 + tid] = s;
  }
  __syncthreads();
  for (int i = tid; i < 32 * 32; i += 256) {
    int n = i >> 5, e = i & 31;
    Hs[i] = Hr[e * 32 + n] / (aux[32 + n] + 1e-8f);
  }
  __syncthreads();
  // out = (H^T/Dv) @ m -> th
  wmma_gemm<32, 128, 32>(Hs, 32, xnm, DD, 0, th, DD, wave, lane);
  __syncthreads();
  float* og = htout + (size_t)b * NN * DD;
  for (int i = tid; i < NN * DD; i += 256) og[i] = fmaxf(th[i], 0.f);
  for (int i = tid; i < NN * NN; i += 256) {
    int e = i / NN, n = i % NN;
    atomicAdd(&cnt[n * NN + e], Hr[e * 32 + n]);
  }
}

// ---------------- shared adjacency from batch counts ----------------
__global__ void k_adj(const float* cnt, float* adj, int N, float thresh)
{
  int idx = blockIdx.x * blockDim.x + threadIdx.x;
  if (idx >= N * N) return;
  int i = idx / N, j = idx % N;
  float a = (i == j) ? 1.f : 0.f;
  for (int e = 0; e < N; ++e)
    a += ((cnt[i * N + e] > thresh) ? 1.f : 0.f) * ((cnt[j * N + e] > thresh) ? 1.f : 0.f);
  adj[idx] = (a > 0.f) ? 1.f : 0.f;
}

// ---------------- stage 3: gat1 + gat2 + time-weight combine ----------------
__global__ __launch_bounds__(256)
void k_gat_time(const float* __restrict__ ht, const float* adj,
                const float* W1, const float* as1, const float* ad1, const float* gb1,
                const float* W2, const float* as2, const float* ad2, const float* gb2,
                const float* tw, float* __restrict__ gout)
{
  __shared__ float x[32 * DD];
  __shared__ float h1[32 * DD];
  __shared__ float o[32 * DD];
  __shared__ float A[32 * 32];
  __shared__ float s[256];
  const int tid = threadIdx.x, lane = tid & 31, wave = tid >> 5;
  const int b = blockIdx.x;
  const float* xg = ht + (size_t)b * NN * DD;

  __builtin_prefetch(W1 + (tid << 6), 0, 0);
  __builtin_prefetch(W2 + (tid << 6), 0, 0);

  for (int i = tid; i < 32 * DD; i += 256) x[i] = (i < NN * DD) ? xg[i] : 0.f;
  for (int i = tid; i < 32 * 32; i += 256) {
    int r = i >> 5, c = i & 31;
    A[i] = (r < NN && c < NN) ? adj[r * NN + c] : 0.f;
  }
  __syncthreads();
  // gat1: heads=2, dh=64
  wmma_gemm<32, 128, 128>(x, DD, W1, DD, 0, h1, DD, wave, lane);
  __syncthreads();
  if (tid < 2 * NN) {
    int i = tid >> 1, hh = tid & 1;
    float ss = 0.f, sd = 0.f;
    for (int d = 0; d < 64; ++d) {
      float v = h1[i * DD + hh * 64 + d];
      ss += v * as1[hh * 64 + d];
      sd += v * ad1[hh * 64 + d];
    }
    s[tid] = ss; s[64 + tid] = sd;
  }
  for (int i = tid; i < 32 * DD; i += 256) o[i] = 0.f;
  __syncthreads();
  if (tid < 2 * NN) {
    int i = tid >> 1, hh = tid & 1;
    float sd = s[64 + tid];
    float mx = -1e30f;
    for (int j = 0; j < NN; ++j)
      if (A[i * 32 + j] > 0.f) mx = fmaxf(mx, lrelu_(sd + s[j * 2 + hh]));
    float sum = 0.f;
    for (int j = 0; j < NN; ++j)
      if (A[i * 32 + j] > 0.f) sum += __expf(lrelu_(sd + s[j * 2 + hh]) - mx);
    float inv = 1.f / sum;
    for (int j = 0; j < NN; ++j)
      if (A[i * 32 + j] > 0.f) {
        float al = __expf(lrelu_(sd + s[j * 2 + hh]) - mx) * inv;
        for (int d = 0; d < 64; ++d)
          o[i * DD + hh * 64 + d] += al * h1[j * DD + hh * 64 + d];
      }
  }
  __syncthreads();
  for (int i = tid; i < NN * DD; i += 256) x[i] = elu_(o[i] + gb1[i & 127]);
  __syncthreads();
  // gat2: heads=1, dh=128
  wmma_gemm<32, 128, 128>(x, DD, W2, DD, 0, h1, DD, wave, lane);
  __syncthreads();
  if (tid < NN) {
    float ss = 0.f, sd = 0.f;
    for (int d = 0; d < DD; ++d) {
      float v = h1[tid * DD + d];
      ss += v * as2[d]; sd += v * ad2[d];
    }
    s[tid] = ss; s[64 + tid] = sd;
  }
  for (int i = tid; i < 32 * DD; i += 256) o[i] = 0.f;
  __syncthreads();
  if (tid < NN) {
    float sd = s[64 + tid];
    float mx = -1e30f;
    for (int j = 0; j < NN; ++j)
      if (A[tid * 32 + j] > 0.f) mx = fmaxf(mx, lrelu_(sd + s[j]));
    float sum = 0.f;
    for (int j = 0; j < NN; ++j)
      if (A[tid * 32 + j] > 0.f) sum += __expf(lrelu_(sd + s[j]) - mx);
    float inv = 1.f / sum;
    for (int j = 0; j < NN; ++j)
      if (A[tid * 32 + j] > 0.f) {
        float al = __expf(lrelu_(sd + s[j]) - mx) * inv;
        for (int d = 0; d < DD; ++d) o[tid * DD + d] += al * h1[j * DD + d];
      }
  }
  __syncthreads();
  float* gg = gout + (size_t)b * 10 * DD;
  for (int idx = tid; idx < 10 * DD; idx += 256) {
    int m = idx >> 7, d = idx & 127;
    float acc = 0.f;
    for (int c = 0; c < 3; ++c)
      acc += elu_(o[(3 * m + c) * DD + d] + gb2[d]) * tw[m * 3 + c];
    gg[idx] = acc;
  }
}

// ---------------- stage 4: hypergraph (10 nodes, k=5) ----------------
__global__ __launch_bounds__(256)
void k_hyper10(const float* __restrict__ gin, const float* W, const float* bias,
               float* __restrict__ g2out, float* cnt)
{
  __shared__ float g[16 * DD];
  __shared__ float th[16 * DD];
  __shared__ float m[16 * DD];
  __shared__ float Ss[16 * 16];
  __shared__ float Hr[16 * 16];
  __shared__ float aux[64];
  const int tid = threadIdx.x, lane = tid & 31, wave = tid >> 5;
  const int b = blockIdx.x;
  const float* xg = gin + (size_t)b * 10 * DD;

  __builtin_prefetch(W + (tid << 6), 0, 0);

  for (int i = tid; i < 16 * DD; i += 256) g[i] = (i < 10 * DD) ? xg[i] : 0.f;
  __syncthreads();
  if (tid < 10) {
    float s = 1e-8f;
    for (int d = 0; d < DD; ++d) { float v = g[tid * DD + d]; s += v * v; }
    aux[32 + tid] = rsqrtf(s);
  }
  __syncthreads();
  if (tid < 100) {
    int i = tid / 10, j = tid % 10;
    float s = 0.f;
    for (int d = 0; d < DD; ++d) s += g[i * DD + d] * g[j * DD + d];
    Ss[i * 16 + j] = s * aux[32 + i] * aux[32 + j];
  }
  __syncthreads();
  if (tid < 16) {
    if (tid < 10) {
      float prev = 1e30f;
      for (int p = 0; p < 5; ++p) {
        float best = -1e30f;
        for (int n = 0; n < 10; ++n) {
          float v = Ss[tid * 16 + n];
          if (v < prev && v > best) best = v;
        }
        prev = best;
      }
      for (int n = 0; n < 16; ++n)
        Hr[tid * 16 + n] = (n < 10 && Ss[tid * 16 + n] >= prev) ? 1.f : 0.f;
    } else {
      for (int n = 0; n < 16; ++n) Hr[tid * 16 + n] = 0.f;
    }
  }
  __syncthreads();
  // theta
  wmma_gemm<16, 128, 128>(g, DD, W, DD, 0, th, DD, wave, lane);
  __syncthreads();
  for (int i = tid; i < 16 * DD; i += 256) th[i] += bias[i & 127];
  if (tid < 16) {
    float de = 0.f, dv = 0.f;
    for (int n = 0; n < 10; ++n) de += Hr[tid * 16 + n];
    for (int e = 0; e < 10; ++e) dv += Hr[e * 16 + tid];
    aux[tid] = de; aux[16 + tid] = dv;
  }
  __syncthreads();
  for (int idx = tid; idx < 10 * DD; idx += 256) {
    int e = idx >> 7, f = idx & 127;
    float acc = 0.f;
    for (int n = 0; n < 10; ++n) acc += Hr[e * 16 + n] * th[n * DD + f];
    m[e * DD + f] = acc / (aux[e] + 1e-8f);
  }
  __syncthreads();
  float* og = g2out + (size_t)b * 10 * DD;
  for (int idx = tid; idx < 10 * DD; idx += 256) {
    int n = idx >> 7, f = idx & 127;
    float acc = 0.f;
    for (int e = 0; e < 10; ++e) acc += Hr[e * 16 + n] * m[e * DD + f];
    og[idx] = fmaxf(acc / (aux[16 + n] + 1e-8f), 0.f);
  }
  if (tid < 100) {
    int e = tid / 10, n = tid % 10;
    atomicAdd(&cnt[n * 10 + e], Hr[e * 16 + n]);
  }
}

// ---------------- stage 5: gat3 + gat4 + mean pool -> out_t ----------------
__global__ __launch_bounds__(256)
void k_gat10(const float* __restrict__ g2, const float* adj,
             const float* W3, const float* as3, const float* ad3, const float* gb3,
             const float* W4, const float* as4, const float* ad4, const float* gb4,
             float* __restrict__ out_t)
{
  __shared__ float x[16 * DD];
  __shared__ float h1[16 * DD];
  __shared__ float o[16 * DD];
  __shared__ float A[16 * 16];
  __shared__ float s[128];
  const int tid = threadIdx.x, lane = tid & 31, wave = tid >> 5;
  const int b = blockIdx.x;
  const float* xg = g2 + (size_t)b * 10 * DD;

  __builtin_prefetch(W3 + (tid << 6), 0, 0);
  __builtin_prefetch(W4 + (tid << 6), 0, 0);

  for (int i = tid; i < 16 * DD; i += 256) x[i] = (i < 10 * DD) ? xg[i] : 0.f;
  for (int i = tid; i < 16 * 16; i += 256) {
    int r = i >> 4, c = i & 15;
    A[i] = (r < 10 && c < 10) ? adj[r * 10 + c] : 0.f;
  }
  __syncthreads();
  wmma_gemm<16, 128, 128>(x, DD, W3, DD, 0, h1, DD, wave, lane);
  __syncthreads();
  if (tid < 20) {
    int i = tid >> 1, hh = tid & 1;
    float ss = 0.f, sd = 0.f;
    for (int d = 0; d < 64; ++d) {
      float v = h1[i * DD + hh * 64 + d];
      ss += v * as3[hh * 64 + d]; sd += v * ad3[hh * 64 + d];
    }
    s[tid] = ss; s[64 + tid] = sd;
  }
  for (int i = tid; i < 16 * DD; i += 256) o[i] = 0.f;
  __syncthreads();
  if (tid < 20) {
    int i = tid >> 1, hh = tid & 1;
    float sd = s[64 + tid];
    float mx = -1e30f;
    for (int j = 0; j < 10; ++j)
      if (A[i * 16 + j] > 0.f) mx = fmaxf(mx, lrelu_(sd + s[j * 2 + hh]));
    float sum = 0.f;
    for (int j = 0; j < 10; ++j)
      if (A[i * 16 + j] > 0.f) sum += __expf(lrelu_(sd + s[j * 2 + hh]) - mx);
    float inv = 1.f / sum;
    for (int j = 0; j < 10; ++j)
      if (A[i * 16 + j] > 0.f) {
        float al = __expf(lrelu_(sd + s[j * 2 + hh]) - mx) * inv;
        for (int d = 0; d < 64; ++d)
          o[i * DD + hh * 64 + d] += al * h1[j * DD + hh * 64 + d];
      }
  }
  __syncthreads();
  for (int i = tid; i < 10 * DD; i += 256) x[i] = elu_(o[i] + gb3[i & 127]);
  __syncthreads();
  wmma_gemm<16, 128, 128>(x, DD, W4, DD, 0, h1, DD, wave, lane);
  __syncthreads();
  if (tid < 10) {
    float ss = 0.f, sd = 0.f;
    for (int d = 0; d < DD; ++d) {
      float v = h1[tid * DD + d];
      ss += v * as4[d]; sd += v * ad4[d];
    }
    s[tid] = ss; s[64 + tid] = sd;
  }
  for (int i = tid; i < 16 * DD; i += 256) o[i] = 0.f;
  __syncthreads();
  if (tid < 10) {
    float sd = s[64 + tid];
    float mx = -1e30f;
    for (int j = 0; j < 10; ++j)
      if (A[tid * 16 + j] > 0.f) mx = fmaxf(mx, lrelu_(sd + s[j]));
    float sum = 0.f;
    for (int j = 0; j < 10; ++j)
      if (A[tid * 16 + j] > 0.f) sum += __expf(lrelu_(sd + s[j]) - mx);
    float inv = 1.f / sum;
    for (int j = 0; j < 10; ++j)
      if (A[tid * 16 + j] > 0.f) {
        float al = __expf(lrelu_(sd + s[j]) - mx) * inv;
        for (int d = 0; d < DD; ++d) o[tid * DD + d] += al * h1[j * DD + d];
      }
  }
  __syncthreads();
  if (tid < DD) {
    float acc = 0.f;
    for (int i = 0; i < 10; ++i) acc += elu_(o[i * DD + tid] + gb4[tid]);
    out_t[(size_t)b * DD + tid] = acc * 0.1f;
  }
}

// ---------------- stage 6: frequency branch (128 nodes, feat 30, 2x hypergraph) ----------------
__global__ __launch_bounds__(256)
void k_freq(const float* __restrict__ hres,
            const float* W11, const float* b11,
            const float* W22, const float* b22,
            float* __restrict__ xf2out, float* cntf)
{
  __shared__ float xf[128 * 32];
  __shared__ float xm[128 * 32];   // xn, then messages m
  __shared__ float th[128 * 32];   // theta
  __shared__ float sc[16 * 128];   // Gram chunk
  __shared__ unsigned Hb[128 * 4]; // membership bitmask (edge-major)
  __shared__ float aux[256];       // norms/De[128], Dv[128]
  __shared__ half_t Wh[32 * 32];

  const int tid = threadIdx.x, lane = tid & 31, wave = tid >> 5;
  const int b = blockIdx.x;
  const float* hg = hres + (size_t)b * NN * DD;

  for (int i = tid; i < 128 * 32; i += 256) {
    int n = i >> 5, f = i & 31;
    xf[i] = (f < NN) ? hg[f * DD + n] : 0.f;
  }
  __syncthreads();

  for (int pass = 0; pass < 2; ++pass) {
    const float* Wp = pass ? W22 : W11;
    const float* bp = pass ? b22 : b11;
    if (tid < 128) {
      float s = 1e-8f;
      for (int f = 0; f < 32; ++f) { float v = xf[tid * 32 + f]; s += v * v; }
      aux[tid] = rsqrtf(s);
    }
    __syncthreads();
    for (int i = tid; i < 128 * 32; i += 256) xm[i] = xf[i] * aux[i >> 5];
    __syncthreads();
    // Gram in 16-row chunks, WMMA transposed-B, top-10 threshold, bitmask
    for (int c = 0; c < 8; ++c) {
      {
        v8f cc = wmma_tile<32>(xm, 32, xm, 32, 1, c * 16, wave * 16, lane);
        int col = wave * 16 + (lane & 15);
        int rb  = (lane >> 4) << 3;
#pragma unroll
        for (int r = 0; r < 8; ++r) sc[(rb + r) * 128 + col] = cc[r];
      }
      __syncthreads();
      if (tid < 16) {
        int e = c * 16 + tid;
        float prev = 1e30f;
        for (int p = 0; p < 10; ++p) {
          float best = -1e30f;
          for (int j = 0; j < 128; ++j) {
            float v = sc[tid * 128 + j];
            if (v < prev && v > best) best = v;
          }
          prev = best;
        }
        unsigned w0 = 0, w1 = 0, w2 = 0, w3 = 0;
        float de = 0.f;
        for (int j = 0; j < 128; ++j)
          if (sc[tid * 128 + j] >= prev) {
            de += 1.f;
            if (j < 32)       w0 |= 1u << j;
            else if (j < 64)  w1 |= 1u << (j - 32);
            else if (j < 96)  w2 |= 1u << (j - 64);
            else              w3 |= 1u << (j - 96);
          }
        Hb[e * 4 + 0] = w0; Hb[e * 4 + 1] = w1; Hb[e * 4 + 2] = w2; Hb[e * 4 + 3] = w3;
        aux[e] = de;   // De (norms no longer needed for these rows)
      }
      __syncthreads();
    }
    // theta = xf @ W (30x30 zero-padded, staged f16)
    for (int i = tid; i < 32 * 32; i += 256) {
      int k = i >> 5, n = i & 31;
      Wh[i] = (half_t)((k < NN && n < NN) ? Wp[k * NN + n] : 0.f);
    }
    __syncthreads();
    wmma_gemm_h<128, 32, 32>(xf, 32, Wh, 32, th, 32, wave, lane);
    __syncthreads();
    for (int i = tid; i < 128 * 32; i += 256) {
      int f = i & 31;
      th[i] += (f < NN) ? bp[f] : 0.f;
    }
    __syncthreads();
    // messages m[e][f]
    for (int idx = tid; idx < 128 * 32; idx += 256) {
      int e = idx >> 5, f = idx & 31;
      float acc = 0.f;
      for (int j = 0; j < 128; ++j)
        if ((Hb[e * 4 + (j >> 5)] >> (j & 31)) & 1u) acc += th[j * 32 + f];
      xm[idx] = acc / (aux[e] + 1e-8f);
    }
    __syncthreads();
    if (tid < 128) {
      float s = 0.f;
      for (int e = 0; e < 128; ++e) s += (float)((Hb[e * 4 + (tid >> 5)] >> (tid & 31)) & 1u);
      aux[128 + tid] = s;   // Dv
    }
    __syncthreads();
    // out[n][f] -> xf (in place, relu)
    for (int idx = tid; idx < 128 * 32; idx += 256) {
      int n = idx >> 5, f = idx & 31;
      float acc = 0.f;
      for (int e = 0; e < 128; ++e)
        if ((Hb[e * 4 + (n >> 5)] >> (n & 31)) & 1u) acc += xm[e * 32 + f];
      xf[idx] = fmaxf(acc / (aux[128 + n] + 1e-8f), 0.f);
    }
    if (pass == 1) {
      for (int idx = tid; idx < 128 * 128; idx += 256) {
        int n = idx >> 7, e = idx & 127;
        if ((Hb[e * 4 + (n >> 5)] >> (n & 31)) & 1u) atomicAdd(&cntf[idx], 1.f);
      }
    }
    __syncthreads();
  }
  float* og = xf2out + (size_t)b * 128 * 32;
  for (int i = tid; i < 128 * 32; i += 256) og[i] = xf[i];
}

// ---------------- stage 7: gat33 + gat44 + mean pool -> out_f ----------------
__global__ __launch_bounds__(256)
void k_gatf(const float* __restrict__ xf2, const float* af,
            const float* W33, const float* as33, const float* ad33, const float* gb33,
            const float* W44, const float* as44, const float* ad44, const float* gb44,
            float* __restrict__ out_f)
{
  __shared__ float x[128 * 32];
  __shared__ float h1[128 * 32];
  __shared__ float o[128 * 32];
  __shared__ float s[512];
  __shared__ half_t Wh[32 * 32];
  const int tid = threadIdx.x, lane = tid & 31, wave = tid >> 5;
  const int b = blockIdx.x;
  const float* xg = xf2 + (size_t)b * 128 * 32;

  for (int i = tid; i < 128 * 32; i += 256) x[i] = xg[i];
  for (int i = tid; i < 32 * 32; i += 256) {
    int k = i >> 5, n = i & 31;
    Wh[i] = (half_t)((k < NN && n < NN) ? W33[k * NN + n] : 0.f);
  }
  __syncthreads();
  wmma_gemm_h<128, 32, 32>(x, 32, Wh, 32, h1, 32, wave, lane);
  __syncthreads();
  {   // heads=2, dh=15 -> one thread per (i, head)
    int i = tid >> 1, hh = tid & 1;
    float ss = 0.f, sd = 0.f;
    for (int d = 0; d < 15; ++d) {
      float v = h1[i * 32 + hh * 15 + d];
      ss += v * as33[hh * 15 + d]; sd += v * ad33[hh * 15 + d];
    }
    s[tid] = ss; s[256 + tid] = sd;
  }
  for (int i = tid; i < 128 * 32; i += 256) o[i] = 0.f;
  __syncthreads();
  {
    int i = tid >> 1, hh = tid & 1;
    const float* arow = af + i * 128;
    float sd = s[256 + tid];
    float mx = -1e30f;
    for (int j = 0; j < 128; ++j)
      if (arow[j] > 0.f) mx = fmaxf(mx, lrelu_(sd + s[j * 2 + hh]));
    float sum = 0.f;
    for (int j = 0; j < 128; ++j)
      if (arow[j] > 0.f) sum += __expf(lrelu_(sd + s[j * 2 + hh]) - mx);
    float inv = 1.f / sum;
    for (int j = 0; j < 128; ++j)
      if (arow[j] > 0.f) {
        float al = __expf(lrelu_(sd + s[j * 2 + hh]) - mx) * inv;
        for (int d = 0; d < 15; ++d)
          o[i * 32 + hh * 15 + d] += al * h1[j * 32 + hh * 15 + d];
      }
  }
  __syncthreads();
  for (int i = tid; i < 128 * 32; i += 256) {
    int f = i & 31;
    x[i] = (f < NN) ? elu_(o[i] + gb33[f]) : 0.f;
  }
  __syncthreads();
  for (int i = tid; i < 32 * 32; i += 256) {
    int k = i >> 5, n = i & 31;
    Wh[i] = (half_t)((k < NN && n < NN) ? W44[k * NN + n] : 0.f);
  }
  __syncthreads();
  wmma_gemm_h<128, 32, 32>(x, 32, Wh, 32, h1, 32, wave, lane);
  __syncthreads();
  if (tid < 128) {
    float ss = 0.f, sd = 0.f;
    for (int d = 0; d < NN; ++d) {
      float v = h1[tid * 32 + d];
      ss += v * as44[d]; sd += v * ad44[d];
    }
    s[tid] = ss; s[256 + tid] = sd;
  }
  for (int i = tid; i < 128 * 32; i += 256) o[i] = 0.f;
  __syncthreads();
  if (tid < 128) {
    const float* arow = af + tid * 128;
    float sd = s[256 + tid];
    float mx = -1e30f;
    for (int j = 0; j < 128; ++j)
      if (arow[j] > 0.f) mx = fmaxf(mx, lrelu_(sd + s[j]));
    float sum = 0.f;
    for (int j = 0; j < 128; ++j)
      if (arow[j] > 0.f) sum += __expf(lrelu_(sd + s[j]) - mx);
    float inv = 1.f / sum;
    for (int j = 0; j < 128; ++j)
      if (arow[j] > 0.f) {
        float al = __expf(lrelu_(sd + s[j]) - mx) * inv;
        for (int d = 0; d < NN; ++d) o[tid * 32 + d] += al * h1[j * 32 + d];
      }
  }
  __syncthreads();
  if (tid < NN) {
    float acc = 0.f;
    for (int i = 0; i < 128; ++i) acc += elu_(o[i * 32 + tid] + gb44[tid]);
    out_f[(size_t)b * NN + tid] = acc * (1.f / 128.f);
  }
}

// ---------------- stage 8: MLP head, batch-tiled WMMA ----------------
__global__ __launch_bounds__(256)
void k_head(const float* __restrict__ out_t, const float* __restrict__ out_f,
            const float* fc1W, const float* fc1b, const float* fc2W, const float* fc2b,
            float* __restrict__ out)
{
  __shared__ float  z[16 * 160];
  __shared__ float  mid[16 * 128];
  __shared__ half_t W1h[160 * 128];   // also reused for fc2 staging
  const int tid = threadIdx.x, lane = tid & 31, wave = tid >> 5;
  const int b0 = blockIdx.x * 16;

  __builtin_prefetch(fc1W + (tid << 6), 0, 0);

  for (int i = tid; i < 16 * 160; i += 256) {
    int r = i / 160, c = i % 160;
    int bb = b0 + r;
    float v = 0.f;
    if (c < 128)       v = out_t[(size_t)bb * 128 + c];
    else if (c < 158)  v = out_f[(size_t)bb * 30 + (c - 128)];
    z[i] = v;
  }
  for (int i = tid; i < 160 * 128; i += 256) {
    int k = i >> 7, n = i & 127;
    W1h[i] = (half_t)((k < 158) ? fc1W[k * 128 + n] : 0.f);
  }
  __syncthreads();
  wmma_gemm_h<16, 128, 160>(z, 160, W1h, 128, mid, 128, wave, lane);
  __syncthreads();
  for (int i = tid; i < 16 * 128; i += 256) mid[i] = fmaxf(mid[i] + fc1b[i & 127], 0.f);
  for (int i = tid; i < 128 * 16; i += 256) {
    int k = i >> 4, n = i & 15;
    W1h[i] = (half_t)((n < 5) ? fc2W[k * 5 + n] : 0.f);
  }
  __syncthreads();
  wmma_gemm_h<16, 16, 128>(mid, 128, W1h, 16, z, 16, wave, lane);
  __syncthreads();
  for (int i = tid; i < 16 * 5; i += 256) {
    int r = i / 5, c = i % 5;
    out[(size_t)(b0 + r) * 5 + c] = z[r * 16 + c] + fc2b[c];
  }
}

// ---------------- workspace layout (float offsets) ----------------
#define WS_H    0u
#define WS_HT   3932160u
#define WS_G    7864320u
#define WS_G2   9175040u
#define WS_OUTT 10485760u
#define WS_XF2  10616832u
#define WS_OUTF 14811136u
#define WS_CNT1 14841856u
#define WS_CNT2 14842880u
#define WS_CNTF 14843008u
#define WS_A1   14859392u
#define WS_A2   14860416u
#define WS_AF   14860544u

extern "C" void kernel_launch(void* const* d_in, const int* in_sizes, int n_in,
                              void* d_out, int out_size, void* d_ws, size_t ws_size,
                              hipStream_t stream)
{
  (void)in_sizes; (void)n_in; (void)out_size; (void)ws_size;
  const float* X     = (const float*)d_in[0];
  const float* cwr   = (const float*)d_in[1];
  const float* cwi   = (const float*)d_in[2];
  const float* ln1g  = (const float*)d_in[3];
  const float* ln1b  = (const float*)d_in[4];
  const float* ln2g  = (const float*)d_in[5];
  const float* ln2b  = (const float*)d_in[6];
  const float* W1    = (const float*)d_in[7];
  const float* b1    = (const float*)d_in[8];
  const float* W2    = (const float*)d_in[9];
  const float* b2    = (const float*)d_in[10];
  const float* fw1   = (const float*)d_in[11];
  const float* fb1   = (const float*)d_in[12];
  const float* fw2   = (const float*)d_in[13];
  const float* fb2   = (const float*)d_in[14];
  const float* lng   = (const float*)d_in[15];
  const float* lnb   = (const float*)d_in[16];
  const float* hg1W  = (const float*)d_in[17];
  const float* hg1b  = (const float*)d_in[18];
  const float* hg2W  = (const float*)d_in[19];
  const float* hg2b  = (const float*)d_in[20];
  const float* hg11W = (const float*)d_in[21];
  const float* hg11b = (const float*)d_in[22];
  const float* hg22W = (const float*)d_in[23];
  const float* hg22b = (const float*)d_in[24];
  const float* g1W   = (const float*)d_in[25];
  const float* g1as  = (const float*)d_in[26];
  const float* g1ad  = (const float*)d_in[27];
  const float* g1b   = (const float*)d_in[28];
  const float* g2W   = (const float*)d_in[29];
  const float* g2as  = (const float*)d_in[30];
  const float* g2ad  = (const float*)d_in[31];
  const float* g2b   = (const float*)d_in[32];
  const float* g3W   = (const float*)d_in[33];
  const float* g3as  = (const float*)d_in[34];
  const float* g3ad  = (const float*)d_in[35];
  const float* g3b   = (const float*)d_in[36];
  const float* g4W   = (const float*)d_in[37];
  const float* g4as  = (const float*)d_in[38];
  const float* g4ad  = (const float*)d_in[39];
  const float* g4b   = (const float*)d_in[40];
  const float* g33W  = (const float*)d_in[41];
  const float* g33as = (const float*)d_in[42];
  const float* g33ad = (const float*)d_in[43];
  const float* g33b  = (const float*)d_in[44];
  const float* g44W  = (const float*)d_in[45];
  const float* g44as = (const float*)d_in[46];
  const float* g44ad = (const float*)d_in[47];
  const float* g44b  = (const float*)d_in[48];
  const float* tw    = (const float*)d_in[49];
  const float* fc1W  = (const float*)d_in[50];
  const float* fc1b  = (const float*)d_in[51];
  const float* fc2W  = (const float*)d_in[52];
  const float* fc2b  = (const float*)d_in[53];

  float* ws   = (float*)d_ws;
  float* h    = ws + WS_H;
  float* ht   = ws + WS_HT;
  float* g    = ws + WS_G;
  float* g2   = ws + WS_G2;
  float* outt = ws + WS_OUTT;
  float* xf2  = ws + WS_XF2;
  float* outf = ws + WS_OUTF;
  float* cnt1 = ws + WS_CNT1;
  float* cnt2 = ws + WS_CNT2;
  float* cntf = ws + WS_CNTF;
  float* a1   = ws + WS_A1;
  float* a2   = ws + WS_A2;
  float* af   = ws + WS_AF;

  // zero batch-shared hyperedge counters (cnt1 | cnt2 | cntf contiguous)
  k_zero<<<(17536 + 255) / 256, 256, 0, stream>>>(cnt1, 17536);

  k_encoder<<<NB, 256, 0, stream>>>(X, cwr, cwi, ln1g, ln1b, ln2g, ln2b,
                                    W1, b1, W2, b2, fw1, fb1, fw2, fb2, lng, lnb, h);
  k_hyper30<<<NB, 256, 0, stream>>>(h, hg1W, hg1b, ht, cnt1);
  k_adj<<<(NN * NN + 255) / 256, 256, 0, stream>>>(cnt1, a1, NN, 512.0f);
  k_gat_time<<<NB, 256, 0, stream>>>(ht, a1, g1W, g1as, g1ad, g1b,
                                     g2W, g2as, g2ad, g2b, tw, g);
  k_hyper10<<<NB, 256, 0, stream>>>(g, hg2W, hg2b, g2, cnt2);
  k_adj<<<1, 256, 0, stream>>>(cnt2, a2, 10, 512.0f);
  k_gat10<<<NB, 256, 0, stream>>>(g2, a2, g3W, g3as, g3ad, g3b,
                                  g4W, g4as, g4ad, g4b, outt);
  k_freq<<<NB, 256, 0, stream>>>(h, hg11W, hg11b, hg22W, hg22b, xf2, cntf);
  k_adj<<<(128 * 128 + 255) / 256, 256, 0, stream>>>(cntf, af, 128, 512.0f);
  k_gatf<<<NB, 256, 0, stream>>>(xf2, af, g33W, g33as, g33ad, g33b,
                                 g44W, g44as, g44ad, g44b, outf);
  k_head<<<NB / 16, 256, 0, stream>>>(outt, outf, fc1W, fc1b, fc2W, fc2b, (float*)d_out);
}
